// CRFDecoder_74440373174921
// MI455X (gfx1250) — 16-bit and 8-bit matrix operand order repaired and checked by reading, compiled, not run on hardware
//
#include <hip/hip_runtime.h>

#define T_DIM 256
#define B_DIM 64
#define H_DIM 1024
#define V_DIM 512
#define NEGV  -10000.0f
#define BOS_ID 1

typedef __attribute__((ext_vector_type(16))) __bf16 v16bf;
typedef __attribute__((ext_vector_type(16))) int    v16i;
typedef __attribute__((ext_vector_type(8)))  float  v8f;

union FragBF { v16bf v; unsigned int u[8];  };
union FragI  { v16i  v; unsigned int u[16]; };
union Acc    { v8f   v; float        f[8];  };

__device__ __forceinline__ unsigned short f2bf(float x) {
  unsigned u = __float_as_uint(x);
  u += 0x7FFFu + ((u >> 16) & 1u);
  return (unsigned short)(u >> 16);
}

__device__ __forceinline__ unsigned char f2e4m3(float x) {
  if (!(x > 0.0f)) return 0;
  x = fminf(x, 448.0f);
  if (x < 0.015625f) {
    int q = (int)(x * 512.0f + 0.5f);
    return (unsigned char)q;
  }
  unsigned u = __float_as_uint(x);
  int e = (int)((u >> 23) & 0xFF) - 127;
  unsigned m3 = ((u & 0x7FFFFF) + (1u << 19)) >> 20;
  if (m3 == 8u) { m3 = 0u; ++e; }
  unsigned r = (unsigned)((e + 7) << 3) | m3;
  if (r >= 0x7Fu) r = 0x7Eu;
  return (unsigned char)r;
}

__device__ __forceinline__ unsigned pack2_e4m3(float a, float b) {
#if __has_builtin(__builtin_amdgcn_cvt_pk_fp8_f32)
  return (unsigned)__builtin_amdgcn_cvt_pk_fp8_f32(a, b, 0, false) & 0xFFFFu;
#else
  return (unsigned)f2e4m3(a) | ((unsigned)f2e4m3(b) << 8);
#endif
}

__device__ __forceinline__ unsigned pack4_e4m3(float a, float b, float c, float d) {
#if __has_builtin(__builtin_amdgcn_cvt_pk_fp8_f32)
  int v = 0;
  v = __builtin_amdgcn_cvt_pk_fp8_f32(a, b, v, false);
  v = __builtin_amdgcn_cvt_pk_fp8_f32(c, d, v, true);
  return (unsigned)v;
#else
  return (unsigned)f2e4m3(a) | ((unsigned)f2e4m3(b) << 8) |
         ((unsigned)f2e4m3(c) << 16) | ((unsigned)f2e4m3(d) << 24);
#endif
}

__global__ __launch_bounds__(256)
void trans_prep_kernel(const float* __restrict__ trans,
                       unsigned char* __restrict__ E8,
                       float* __restrict__ rowmax) {
  __shared__ float red[256];
  const int i = blockIdx.x, tid = threadIdx.x;
  float m = -3.4e38f;
  for (int j = tid; j < V_DIM; j += 256) m = fmaxf(m, trans[i * V_DIM + j]);
  red[tid] = m; __syncthreads();
  for (int o = 128; o > 0; o >>= 1) {
    if (tid < o) red[tid] = fmaxf(red[tid], red[tid + o]);
    __syncthreads();
  }
  const float rm = red[0];
  {
    int j0 = tid * 2;
    float x0 = __expf(trans[i * V_DIM + j0 + 0] - rm);
    float x1 = __expf(trans[i * V_DIM + j0 + 1] - rm);
    *(unsigned short*)&E8[i * V_DIM + j0] = (unsigned short)pack2_e4m3(x0, x1);
  }
  if (tid == 0) rowmax[i] = rm;
}

__global__ __launch_bounds__(128)
void emission_kernel(const float* __restrict__ enc, const float* __restrict__ W,
                     const float* __restrict__ bias, float* __restrict__ feats) {
  __shared__ unsigned short ldsA[64 * 36];
  __shared__ unsigned short ldsB[64 * 36];
  const int m0 = blockIdx.x * 64;
  const int n0 = blockIdx.y * 64;
  const int tid  = threadIdx.x;
  const int wave = tid >> 5, lane = tid & 31;
  const int ln = lane & 15, hi = lane >> 4;

  Acc acc[4];
  for (int s = 0; s < 4; ++s)
    for (int r = 0; r < 8; ++r) acc[s].f[r] = 0.0f;

  for (int kk = 0; kk < H_DIM; kk += 32) {
    __syncthreads();
    for (int e = 0; e < 16; ++e) {
      int idx = e * 128 + tid, r = idx >> 5, c = idx & 31;
      ldsA[r * 36 + c] = f2bf(enc[(size_t)(m0 + r) * H_DIM + kk + c]);
      ldsB[r * 36 + c] = f2bf(W[(size_t)(n0 + r) * H_DIM + kk + c]);
    }
    __syncthreads();

    FragBF a;
#pragma unroll
    for (int v = 0; v < 8; ++v) {
      int k0 = (v < 4 ? 2 * v : 16 + 2 * (v - 4)) + (hi ? 8 : 0);
      a.u[v] = *(const unsigned int*)&ldsA[(wave * 16 + ln) * 36 + k0];
    }
#pragma unroll
    for (int s = 0; s < 4; ++s) {
      FragBF b;
#pragma unroll
      for (int v = 0; v < 8; ++v) {
        int k0 = (v < 4 ? 2 * v : 16 + 2 * (v - 4)) + (hi ? 8 : 0);
        b.u[v] = *(const unsigned int*)&ldsB[(s * 16 + ln) * 36 + k0];
      }
      acc[s].v = __builtin_amdgcn_wmma_f32_16x16x32_bf16(
          false, a.v, false, b.v, (short)0, acc[s].v, false, false);
    }
  }
  for (int s = 0; s < 4; ++s) {
    int col = n0 + s * 16 + ln;
    float bv = bias[col];
#pragma unroll
    for (int r = 0; r < 8; ++r) {
      int m = m0 + wave * 16 + (hi ? 8 : 0) + r;
      feats[(size_t)m * V_DIM + col] = acc[s].f[r] + bv;
    }
  }
}

#define E_STRIDE   528
#define P_STRIDE   528
#define SM_SCORE   0
#define SM_ROWM    (16 * 512 * 4)
#define SM_MB      (SM_ROWM + 512 * 4)
#define SM_LEN     (SM_MB + 16 * 4)
#define SM_P       (SM_LEN + 16 * 4)
#define SM_E       (SM_P + 16 * P_STRIDE)
#define SM_TOTAL   (SM_E + 512 * E_STRIDE)

__global__ __launch_bounds__(256)
void scan_kernel(const float* __restrict__ feats, const unsigned char* __restrict__ E8,
                 const float* __restrict__ rowmax, const int* __restrict__ lengths,
                 float* __restrict__ out) {
  extern __shared__ char smem[];
  float*         sScore = (float*)(smem + SM_SCORE);
  float*         sRowm  = (float*)(smem + SM_ROWM);
  float*         sMb    = (float*)(smem + SM_MB);
  int*           sLen   = (int*)  (smem + SM_LEN);
  unsigned char* sP     = (unsigned char*)(smem + SM_P);
  unsigned char* sE     = (unsigned char*)(smem + SM_E);

  const int wg = blockIdx.x;
  const int tid = threadIdx.x;
  const int wave = tid >> 5, lane = tid & 31, ln = lane & 15, hi = lane >> 4;

  for (int i = tid; i < V_DIM; i += 256) sRowm[i] = rowmax[i];
  if (tid < 16) sLen[tid] = lengths[wg * 16 + tid];
  for (int i = tid; i < 16 * V_DIM; i += 256) {
    int c = i & 511;
    sScore[i] = (c == BOS_ID) ? 0.0f : NEGV;
  }
  {
    const unsigned long long gbase = (unsigned long long)E8;
    for (int it = tid; it < 512 * 32; it += 256) {
      int r = it >> 5, c = it & 31;
      unsigned voff    = (unsigned)(r * 512 + c * 16);
      unsigned ldsaddr = (unsigned)(unsigned long long)(sE + r * E_STRIDE + c * 16);
      asm volatile("global_load_async_to_lds_b128 %0, %1, %2"
                   :: "v"(ldsaddr), "v"(voff), "s"(gbase) : "memory");
    }
    asm volatile("s_wait_asynccnt 0" ::: "memory");
  }
  __syncthreads();

  const int row = tid >> 4;
  const int c16 = tid & 15;

  for (int t = 0; t < T_DIM; ++t) {
    float m = -3.4e38f;
    for (int j = c16; j < V_DIM; j += 16) m = fmaxf(m, sScore[row * V_DIM + j]);
#pragma unroll
    for (int o = 8; o >= 1; o >>= 1) m = fmaxf(m, __shfl_xor(m, o, 32));
    if (c16 == 0) sMb[row] = m;
    {
      const float* sr = sScore + row * V_DIM + c16 * 32;
      unsigned* dst = (unsigned*)(sP + row * P_STRIDE + c16 * 32);
#pragma unroll
      for (int jj = 0; jj < 32; jj += 4) {
        float x0 = __expf(sr[jj + 0] - m), x1 = __expf(sr[jj + 1] - m);
        float x2 = __expf(sr[jj + 2] - m), x3 = __expf(sr[jj + 3] - m);
        dst[jj >> 2] = pack4_e4m3(x0, x1, x2, x3);
      }
    }
    __syncthreads();

    if (t + 1 < T_DIM)
      __builtin_prefetch(&feats[((size_t)(t + 1) * B_DIM + wg * 16 + hi * 8) * V_DIM
                                + wave * 64 + ln], 0, 1);

    Acc acc[4];
    for (int s = 0; s < 4; ++s)
#pragma unroll
      for (int r = 0; r < 8; ++r) acc[s].f[r] = 0.0f;

    for (int kk = 0; kk < V_DIM; kk += 128) {
      FragI a;
#pragma unroll
      for (int v = 0; v < 16; ++v) {
        int blk = v >> 3, vv = v & 7, p = vv >> 1, sub = vv & 1;
        int k0 = kk + 64 * blk + 16 * p + (hi ? 8 : 0) + 4 * sub;
        a.u[v] = *(const unsigned int*)&sP[ln * P_STRIDE + k0];
      }
#pragma unroll
      for (int s = 0; s < 4; ++s) {
        int n = wave * 64 + s * 16 + ln;
        FragI b;
#pragma unroll
        for (int v = 0; v < 16; ++v) {
          int k0 = kk + 64 * (v >> 3) + 16 * ((v & 7) >> 1) + (hi ? 8 : 0) + 4 * (v & 1);
          b.u[v] = *(const unsigned int*)&sE[n * E_STRIDE + k0];
        }
        acc[s].v = __builtin_amdgcn_wmma_f32_16x16x128_fp8_fp8(
            a.v, b.v, (short)0, acc[s].v, false, false);
      }
    }

#pragma unroll
    for (int s = 0; s < 4; ++s) {
      int i = wave * 64 + s * 16 + ln;
      float rm = sRowm[i];
#pragma unroll
      for (int r = 0; r < 8; ++r) {
        int M = (hi ? 8 : 0) + r;
        float nv = sMb[M] + rm + __logf(fmaxf(acc[s].f[r], 1e-30f))
                 + feats[((size_t)t * B_DIM + wg * 16 + M) * V_DIM + i];
        if (t < sLen[M]) sScore[M * V_DIM + i] = nv;
      }
    }
    __syncthreads();
  }

  float m = -3.4e38f;
  for (int j = c16; j < V_DIM; j += 16) m = fmaxf(m, sScore[row * V_DIM + j]);
#pragma unroll
  for (int o = 8; o >= 1; o >>= 1) m = fmaxf(m, __shfl_xor(m, o, 32));
  float ssum = 0.0f;
  for (int j = c16; j < V_DIM; j += 16) ssum += __expf(sScore[row * V_DIM + j] - m);
#pragma unroll
  for (int o = 8; o >= 1; o >>= 1) ssum += __shfl_xor(ssum, o, 32);
  if (c16 == 0) out[wg * 16 + row] = m + __logf(ssum);
}

extern "C" void kernel_launch(void* const* d_in, const int* in_sizes, int n_in,
                              void* d_out, int out_size, void* d_ws, size_t ws_size,
                              hipStream_t stream) {
  (void)in_sizes; (void)n_in; (void)out_size; (void)ws_size;
  const float* enc     = (const float*)d_in[0];
  const int*   lengths = (const int*)  d_in[1];
  const float* W       = (const float*)d_in[2];
  const float* bias    = (const float*)d_in[3];
  const float* trans   = (const float*)d_in[4];
  float* out = (float*)d_out;

  float*         feats = (float*)d_ws;
  unsigned char* E8    = (unsigned char*)d_ws + (size_t)T_DIM * B_DIM * V_DIM * 4;
  float*         rmax  = (float*)(E8 + (size_t)V_DIM * V_DIM);

  trans_prep_kernel<<<V_DIM, 256, 0, stream>>>(trans, E8, rmax);
  emission_kernel<<<dim3((T_DIM * B_DIM) / 64, V_DIM / 64), 128, 0, stream>>>(
      enc, W, bias, feats);
  scan_kernel<<<B_DIM / 16, 256, SM_TOTAL, stream>>>(feats, E8, rmax, lengths, out);
}
